// PointNetFeaturePropagation_4080218931822
// MI455X (gfx1250) — matrix operand order repaired and checked by reading, not compiled, not run on hardware
//
#include <hip/hip_runtime.h>

// ---------------------------------------------------------------------------
// PointNet Feature Propagation for MI455X (gfx1250, wave32, WMMA)
//
//  Stage 1: 3-NN (scalar-load candidate table, early-out insert) +
//           inverse-distance interp -> xcat bf16 [65536, 384]
//  Stage 2: 3x { WMMA bf16 GEMM with fused deterministic BN partial stats ;
//                two-stage stats reduce ; BN+ReLU }
//
//  GEMMs: V_WMMA_F32_16X16X32_BF16, wave owns a 16x128 strip (8 f32 acc
//  tiles, A-fragment reused 8x). Conv bias dropped (cancels under BN).
//  All reductions are fixed-order -> bitwise deterministic across replays.
// ---------------------------------------------------------------------------

typedef __attribute__((ext_vector_type(16))) __bf16 v16bf;
typedef __attribute__((ext_vector_type(8)))  __bf16 v8bf;
typedef __attribute__((ext_vector_type(8)))  float  v8f;

#define B_   4
#define N_   16384
#define M_   4096
#define C1_  128
#define C2_  256
#define INC_ 384
#define BN_  (B_ * N_)
#define MBLK_ (BN_ / 64)   // 1024 GEMM row-blocks (4 waves x 16 rows each)

// ---------------------------- f32 -> bf16 convert --------------------------
__global__ __launch_bounds__(256) void cvt_f32_bf16(const float* __restrict__ in,
                                                    __bf16* __restrict__ out, int n) {
  int i = blockIdx.x * 256 + threadIdx.x;
  if (i < n) out[i] = (__bf16)in[i];
}

// --------------- build (x, y, z, |p|^2) candidate table --------------------
__global__ __launch_bounds__(256) void build_p4(const float* __restrict__ xyz2,
                                                float4* __restrict__ t4) {
  int i = blockIdx.x * 256 + threadIdx.x;      // b*M_ + m
  if (i < B_ * M_) {
    float x = xyz2[3 * i + 0], y = xyz2[3 * i + 1], z = xyz2[3 * i + 2];
    t4[i] = make_float4(x, y, z, fmaf(x, x, fmaf(y, y, z * z)));
  }
}

// ------------------------- 3-NN + interpolation ----------------------------
// grid: (N_/256, B_), block 256. Candidate reads are wave-uniform -> SMEM.
__global__ __launch_bounds__(256) void knn_interp(const float* __restrict__ xyz1,
                                                  const float* __restrict__ pts1,
                                                  const float* __restrict__ pts2,
                                                  const float4* __restrict__ t4,
                                                  __bf16* __restrict__ xcat) {
  __shared__ int   sidx[256 * 3];
  __shared__ float swt [256 * 3];

  const int b     = blockIdx.y;
  const int nbase = blockIdx.x * 256;
  const int n     = nbase + threadIdx.x;

  const size_t q = ((size_t)b * N_ + n) * 3;
  const float qx = xyz1[q + 0], qy = xyz1[q + 1], qz = xyz1[q + 2];
  const float qq  = fmaf(qx, qx, fmaf(qy, qy, qz * qz));
  const float m2x = -2.0f * qx, m2y = -2.0f * qy, m2z = -2.0f * qz;

  const float4* __restrict__ tb = t4 + (size_t)b * M_;   // uniform-index reads

  float d0 = 3.0e38f, d1 = 3.0e38f, d2 = 3.0e38f;
  int   i0 = 0, i1 = 0, i2 = 0;
#pragma unroll 4
  for (int m = 0; m < M_; ++m) {
    float4 p = tb[m];                                    // s_load (uniform)
    // shifted distance: |p|^2 - 2 q.p  (true d2 = this + |q|^2, order same)
    float d = fmaf(m2x, p.x, fmaf(m2y, p.y, fmaf(m2z, p.z, p.w)));
    if (d < d2) {                                        // rare after warmup
      if (d < d0)      { d2 = d1; i2 = i1; d1 = d0; i1 = i0; d0 = d; i0 = m; }
      else if (d < d1) { d2 = d1; i2 = i1; d1 = d;  i1 = m; }
      else             { d2 = d;  i2 = m; }
    }
  }
  float w0 = 1.0f / (d0 + qq + 1e-8f);
  float w1 = 1.0f / (d1 + qq + 1e-8f);
  float w2 = 1.0f / (d2 + qq + 1e-8f);
  float wr = 1.0f / (w0 + w1 + w2);
  sidx[threadIdx.x * 3 + 0] = i0;  swt[threadIdx.x * 3 + 0] = w0 * wr;
  sidx[threadIdx.x * 3 + 1] = i1;  swt[threadIdx.x * 3 + 1] = w1 * wr;
  sidx[threadIdx.x * 3 + 2] = i2;  swt[threadIdx.x * 3 + 2] = w2 * wr;
  __syncthreads();

  // phase 2: channels across lanes -> coalesced gathers of pts2 rows
  const float* p2 = pts2 + (size_t)b * M_ * C2_;
  const int c = threadIdx.x;                  // 0..255
  for (int p = 0; p < 256; ++p) {
    const size_t g  = (size_t)b * N_ + nbase + p;
    const int j0 = sidx[p * 3 + 0], j1 = sidx[p * 3 + 1], j2 = sidx[p * 3 + 2];
    const float a0 = swt[p * 3 + 0], a1 = swt[p * 3 + 1], a2 = swt[p * 3 + 2];
    float v = a0 * p2[(size_t)j0 * C2_ + c]
            + a1 * p2[(size_t)j1 * C2_ + c]
            + a2 * p2[(size_t)j2 * C2_ + c];
    xcat[g * INC_ + C1_ + c] = (__bf16)v;
    if (c < C1_) xcat[g * INC_ + c] = (__bf16)pts1[g * C1_ + c];
  }
}

// ------------------------------- WMMA GEMM ---------------------------------
// H[BN, ldc] = A[BN, K](bf16) * W[Cout, K]^T(bf16), f32 accumulate, plus
// deterministic per-block BN partials (sum, sumsq per column).
template <int K>
__global__ __launch_bounds__(128) void gemm_bf16_wmma(const __bf16* __restrict__ A,
                                                      const __bf16* __restrict__ W,
                                                      float* __restrict__ H, int ldc,
                                                      float* __restrict__ psum,
                                                      float* __restrict__ psumsq) {
  __shared__ float sS[8][128];
  __shared__ float sQ[8][128];

  const int lane = threadIdx.x & 31;
  const int wave = threadIdx.x >> 5;
  const int m0   = (blockIdx.x * 4 + wave) * 16;
  const int n0   = blockIdx.y * 128;
  const int l16  = lane & 15;
  const int hi   = lane >> 4;

  v8f acc[8];
#pragma unroll
  for (int t = 0; t < 8; ++t) acc[t] = (v8f)(0.0f);

  const __bf16* aRow = A + (size_t)(m0 + l16) * K + hi * 16;
  const __bf16* bRow = W + (size_t)(n0 + l16) * K + hi * 16;

  for (int k0 = 0; k0 < K; k0 += 32) {
    v8bf alo = *(const v8bf*)(aRow + k0);
    v8bf ahi = *(const v8bf*)(aRow + k0 + 8);
    v16bf a = __builtin_shufflevector(alo, ahi, 0, 1, 2, 3, 4, 5, 6, 7,
                                                8, 9, 10, 11, 12, 13, 14, 15);
    v16bf bfrag[8];
#pragma unroll
    for (int t = 0; t < 8; ++t)       // one big load clause
      bfrag[t] = *(const v16bf*)(bRow + (size_t)t * 16 * K + k0);
#pragma unroll
    for (int t = 0; t < 8; ++t)       // WMMA burst, loadcnt counts down
      acc[t] = __builtin_amdgcn_wmma_f32_16x16x32_bf16(
          false, a, false, bfrag[t], (short)0, acc[t], false, false);
  }

  // store D tiles + per-lane column partials
#pragma unroll
  for (int t = 0; t < 8; ++t) {
    float s = 0.f, sq = 0.f;
#pragma unroll
    for (int r = 0; r < 8; ++r) {
      const float v = acc[t][r];
      H[(size_t)(m0 + r + hi * 8) * ldc + (n0 + t * 16 + l16)] = v;
      s += v;
      sq = fmaf(v, v, sq);
    }
    sS[wave * 2 + hi][t * 16 + l16] = s;
    sQ[wave * 2 + hi][t * 16 + l16] = sq;
  }
  __syncthreads();

  if (threadIdx.x < 128) {            // fixed-order block reduction
    const int c = threadIdx.x;
    float s = 0.f, sq = 0.f;
#pragma unroll
    for (int j = 0; j < 8; ++j) { s += sS[j][c]; sq += sQ[j][c]; }
    const int pblk = blockIdx.y * gridDim.x + blockIdx.x;
    psum  [(size_t)pblk * 128 + c] = s;
    psumsq[(size_t)pblk * 128 + c] = sq;
  }
}

// -------------------- deterministic stats reduction ------------------------
// stage 1: grid (16, strips), block 128; sums 64 of the 1024 row-blocks
__global__ __launch_bounds__(128) void stats_reduce1(const float* __restrict__ psum,
                                                     const float* __restrict__ psumsq,
                                                     float* __restrict__ inter) {
  const int c = threadIdx.x;
  const int strip = blockIdx.y;
  const int j = blockIdx.x;           // 0..15
  float s = 0.f, sq = 0.f;
  for (int i = j * 64; i < (j + 1) * 64; ++i) {
    const size_t p = ((size_t)strip * MBLK_ + i) * 128 + c;
    s += psum[p];
    sq += psumsq[p];
  }
  inter[((strip * 16 + j) * 2 + 0) * 128 + c] = s;
  inter[((strip * 16 + j) * 2 + 1) * 128 + c] = sq;
}

// stage 2: 1 block, C threads
template <int C>
__global__ void stats_reduce2(const float* __restrict__ inter,
                              float* __restrict__ stats) {
  const int c = threadIdx.x;
  const int strip = c >> 7, col = c & 127;
  float s = 0.f, sq = 0.f;
  for (int j = 0; j < 16; ++j) {
    s  += inter[((strip * 16 + j) * 2 + 0) * 128 + col];
    sq += inter[((strip * 16 + j) * 2 + 1) * 128 + col];
  }
  stats[c]     = s;
  stats[C + c] = sq;
}

// ------------------------------ BN + ReLU ----------------------------------
template <int C, bool BF16OUT>
__global__ void bn_relu(const float* __restrict__ H, const float* __restrict__ stats,
                        const float* __restrict__ gamma, const float* __restrict__ beta,
                        void* __restrict__ outv, int rowsPerBlk) {
  const int c = threadIdx.x;
  const float inv  = 1.0f / (float)BN_;
  const float mean = stats[c] * inv;
  const float var  = stats[C + c] * inv - mean * mean;
  const float sc   = gamma[c] * rsqrtf(var + 1e-5f);
  const float sh   = beta[c] - mean * sc;
  const int r0 = blockIdx.x * rowsPerBlk;
  for (int r = 0; r < rowsPerBlk; ++r) {
    float v = fmaf(H[(size_t)(r0 + r) * C + c], sc, sh);
    v = v > 0.f ? v : 0.f;
    if (BF16OUT) ((__bf16*)outv)[(size_t)(r0 + r) * C + c] = (__bf16)v;
    else         ((float*) outv)[(size_t)(r0 + r) * C + c] = v;
  }
}

// ------------------------------ orchestration ------------------------------
static const size_t OFF_WB0   = 0;                                      // 256*384 bf16
static const size_t OFF_WB1   = OFF_WB0   + (size_t)256 * 384 * 2;
static const size_t OFF_WB2   = OFF_WB1   + (size_t)256 * 256 * 2;
static const size_t OFF_T4    = OFF_WB2   + (size_t)128 * 256 * 2;      // B*M float4
static const size_t OFF_XCAT  = OFF_T4    + (size_t)B_ * M_ * 16;       // BN*384 bf16
static const size_t OFF_XA    = OFF_XCAT  + (size_t)BN_ * INC_ * 2;     // BN*256 bf16
static const size_t OFF_H     = OFF_XA    + (size_t)BN_ * 256 * 2;      // BN*256 f32
static const size_t OFF_PS    = OFF_H     + (size_t)BN_ * 256 * 4;      // 2*1024*128 f32
static const size_t OFF_PSQ   = OFF_PS    + (size_t)2 * MBLK_ * 128 * 4;
static const size_t OFF_INTER = OFF_PSQ   + (size_t)2 * MBLK_ * 128 * 4; // 32KB
static const size_t OFF_ST    = OFF_INTER + (size_t)2 * 16 * 2 * 128 * 4;

extern "C" void kernel_launch(void* const* d_in, const int* in_sizes, int n_in,
                              void* d_out, int out_size, void* d_ws, size_t ws_size,
                              hipStream_t stream) {
  const float* xyz1 = (const float*)d_in[0];
  const float* xyz2 = (const float*)d_in[1];
  const float* pts1 = (const float*)d_in[2];
  const float* pts2 = (const float*)d_in[3];
  const float* w0   = (const float*)d_in[4];
  const float* g0   = (const float*)d_in[6];
  const float* be0  = (const float*)d_in[7];
  const float* w1   = (const float*)d_in[8];
  const float* g1   = (const float*)d_in[10];
  const float* be1  = (const float*)d_in[11];
  const float* w2   = (const float*)d_in[12];
  const float* g2   = (const float*)d_in[14];
  const float* be2  = (const float*)d_in[15];
  (void)in_sizes; (void)n_in; (void)out_size; (void)ws_size;

  char* ws = (char*)d_ws;
  __bf16* wb0   = (__bf16*)(ws + OFF_WB0);
  __bf16* wb1   = (__bf16*)(ws + OFF_WB1);
  __bf16* wb2   = (__bf16*)(ws + OFF_WB2);
  float4* t4    = (float4*)(ws + OFF_T4);
  __bf16* xcat  = (__bf16*)(ws + OFF_XCAT);
  __bf16* xa    = (__bf16*)(ws + OFF_XA);
  float*  H     = (float*)(ws + OFF_H);
  float*  ps    = (float*)(ws + OFF_PS);
  float*  psq   = (float*)(ws + OFF_PSQ);
  float*  inter = (float*)(ws + OFF_INTER);
  float*  st    = (float*)(ws + OFF_ST);

  // prep: weight conversion (bias cancels under training BN) + |p|^2 table
  cvt_f32_bf16<<<(256 * 384 + 255) / 256, 256, 0, stream>>>(w0, wb0, 256 * 384);
  cvt_f32_bf16<<<(256 * 256 + 255) / 256, 256, 0, stream>>>(w1, wb1, 256 * 256);
  cvt_f32_bf16<<<(128 * 256 + 255) / 256, 256, 0, stream>>>(w2, wb2, 128 * 256);
  build_p4<<<(B_ * M_ + 255) / 256, 256, 0, stream>>>(xyz2, t4);

  // stage 1: 3-NN + interpolation -> xcat [BN, 384] bf16
  knn_interp<<<dim3(N_ / 256, B_), 256, 0, stream>>>(xyz1, pts1, pts2, t4, xcat);

  const int RPB = BN_ / 256;

  // layer 0: 384 -> 256
  gemm_bf16_wmma<384><<<dim3(MBLK_, 2), 128, 0, stream>>>(xcat, wb0, H, 256, ps, psq);
  stats_reduce1<<<dim3(16, 2), 128, 0, stream>>>(ps, psq, inter);
  stats_reduce2<256><<<1, 256, 0, stream>>>(inter, st);
  bn_relu<256, true><<<256, 256, 0, stream>>>(H, st, g0, be0, xa, RPB);

  // layer 1: 256 -> 256
  gemm_bf16_wmma<256><<<dim3(MBLK_, 2), 128, 0, stream>>>(xa, wb1, H, 256, ps, psq);
  stats_reduce1<<<dim3(16, 2), 128, 0, stream>>>(ps, psq, inter);
  stats_reduce2<256><<<1, 256, 0, stream>>>(inter, st);
  bn_relu<256, true><<<256, 256, 0, stream>>>(H, st, g1, be1, xa, RPB);

  // layer 2: 256 -> 128 (f32 out)
  gemm_bf16_wmma<256><<<dim3(MBLK_, 1), 128, 0, stream>>>(xa, wb2, H, 128, ps, psq);
  stats_reduce1<<<dim3(16, 1), 128, 0, stream>>>(ps, psq, inter);
  stats_reduce2<128><<<1, 128, 0, stream>>>(inter, st);
  bn_relu<128, false><<<256, 128, 0, stream>>>(H, st, g2, be2, (float*)d_out, RPB);
}
